// AttentionMatrix_41841571398230
// MI455X (gfx1250) — matrix operand order repaired and checked by reading, not compiled, not run on hardware
//
#include <hip/hip_runtime.h>

// ---- vector types ----
typedef float  v4f   __attribute__((ext_vector_type(4)));
typedef float  v8f   __attribute__((ext_vector_type(8)));
typedef __bf16 v4bf  __attribute__((ext_vector_type(4)));
typedef __bf16 v8bf  __attribute__((ext_vector_type(8)));
typedef __bf16 v16bf __attribute__((ext_vector_type(16)));

#define BM 128          // block tile rows (n)
#define BN 128          // block tile cols (m)
#define BK 32           // fp32 k-slab per iteration (= bf16 WMMA K)
#define LDSS 48         // LDS row stride in bf16 elems (96B: 16B-aligned, low-conflict)
#define NT 256          // threads per block (8 wave32)

// Combine two 16B LDS loads into one v16bf fragment.
__device__ __forceinline__ v16bf frag2(const __bf16* p0, const __bf16* p1) {
    v8bf a = *(const v8bf*)p0;
    v8bf b = *(const v8bf*)p1;
    return __builtin_shufflevector(a, b, 0,1,2,3,4,5,6,7,8,9,10,11,12,13,14,15);
}

__global__ __launch_bounds__(NT)
void attn_qkt_bf16x3_kernel(const float* __restrict__ A,   // [b, N, E]
                            const float* __restrict__ B,   // [b, M, E]
                            float* __restrict__ C,         // [b, N, M]
                            int Nrows, int Mcols, int E) {
    // Double-buffered LDS: 4 planes x 2 buffers x 12 KB = 96 KB
    __shared__ __bf16 sAhi[2][BM * LDSS];
    __shared__ __bf16 sAlo[2][BM * LDSS];
    __shared__ __bf16 sBhi[2][BN * LDSS];
    __shared__ __bf16 sBlo[2][BN * LDSS];

    const int t      = threadIdx.x;
    const int lane   = t & 31;
    const int wave   = t >> 5;
    const int waveM  = wave & 1;   // 2 waves along M: 64 rows each
    const int waveN  = wave >> 1;  // 4 waves along N: 32 cols each
    const int r16    = lane & 15;
    const int laneHi = lane >> 4;  // 0 or 1

    const int nBase = blockIdx.y * BM;
    const int mBase = blockIdx.x * BN;
    const int batch = blockIdx.z;

    const float* Ab = A + (size_t)batch * Nrows * E;
    const float* Bb = B + (size_t)batch * Mcols * E;

    // Staging map: 256 threads x 4 v4f cover a 128x32 fp32 tile.
    int rowf[4], colf[4];
#pragma unroll
    for (int j = 0; j < 4; ++j) {
        int f = t + NT * j;        // [0,1024)
        rowf[j] = f >> 3;          // 8 v4f per 32-float row
        colf[j] = (f & 7) << 2;
    }

    v8f acc[4][2];
#pragma unroll
    for (int tm = 0; tm < 4; ++tm)
#pragma unroll
        for (int tn = 0; tn < 2; ++tn)
            acc[tm][tn] = (v8f)(0.0f);

    const int NK = E / BK;
    v4f aReg[4], bReg[4];

    auto gload = [&](int k0) {
#pragma unroll
        for (int j = 0; j < 4; ++j) {
            aReg[j] = *(const v4f*)(Ab + (size_t)(nBase + rowf[j]) * E + k0 + colf[j]);
            bReg[j] = *(const v4f*)(Bb + (size_t)(mBase + rowf[j]) * E + k0 + colf[j]);
        }
    };

    // Packed fp32 -> bf16 hi/lo split (cvt_pk + pk-sub), store to LDS buffer.
    auto convert_store = [&](int buf) {
#pragma unroll
        for (int j = 0; j < 4; ++j) {
            v4f  a  = aReg[j];
            v4bf ah = __builtin_convertvector(a, v4bf);
            v4f  ar = a - __builtin_convertvector(ah, v4f);
            v4bf al = __builtin_convertvector(ar, v4bf);

            v4f  b  = bReg[j];
            v4bf bh = __builtin_convertvector(b, v4bf);
            v4f  br = b - __builtin_convertvector(bh, v4f);
            v4bf bl = __builtin_convertvector(br, v4bf);

            int off = rowf[j] * LDSS + colf[j];
            *(v4bf*)(sAhi[buf] + off) = ah;
            *(v4bf*)(sAlo[buf] + off) = al;
            *(v4bf*)(sBhi[buf] + off) = bh;
            *(v4bf*)(sBlo[buf] + off) = bl;
        }
    };

    auto compute = [&](int buf) {
        // B fragments (32x16, lane=N; K chunks {0..7,16..23} / {8..15,24..31})
        v16bf bfh[2], bfl[2];
#pragma unroll
        for (int tn = 0; tn < 2; ++tn) {
            int base = (waveN * 32 + tn * 16 + r16) * LDSS;
            int ko   = (laneHi << 3);                       // 0 or 8
            bfh[tn] = frag2(sBhi[buf] + base + ko, sBhi[buf] + base + ko + 16);
            bfl[tn] = frag2(sBlo[buf] + base + ko, sBlo[buf] + base + ko + 16);
        }
        // A fragments (16x32, lane=M; K chunks {0..7,16..23} / {8..15,24..31})
#pragma unroll
        for (int tm = 0; tm < 4; ++tm) {
            int base = (waveM * 64 + tm * 16 + r16) * LDSS;
            int ko   = (laneHi << 3);                       // 0 or 8
            v16bf afh = frag2(sAhi[buf] + base + ko, sAhi[buf] + base + ko + 16);
            v16bf afl = frag2(sAlo[buf] + base + ko, sAlo[buf] + base + ko + 16);
#pragma unroll
            for (int tn = 0; tn < 2; ++tn) {
                acc[tm][tn] = __builtin_amdgcn_wmma_f32_16x16x32_bf16(
                    false, afh, false, bfh[tn], (short)0, acc[tm][tn], false, false);
                acc[tm][tn] = __builtin_amdgcn_wmma_f32_16x16x32_bf16(
                    false, afh, false, bfl[tn], (short)0, acc[tm][tn], false, false);
                acc[tm][tn] = __builtin_amdgcn_wmma_f32_16x16x32_bf16(
                    false, afl, false, bfh[tn], (short)0, acc[tm][tn], false, false);
            }
        }
    };

    // ---- pipeline prologue ----
    gload(0);
    convert_store(0);

    // ---- main loop: ONE barrier per iteration (double-buffered LDS) ----
    for (int kk = 0; kk < NK; ++kk) {
        const int cur = kk & 1;
        __syncthreads();                 // buffer `cur` visible to all waves
        if (kk + 1 < NK) gload((kk + 1) * BK);   // VMEM in flight across compute
        compute(cur);
        if (kk + 1 < NK) convert_store(1 - cur); // waits loads; writes `nxt` only
    }

    // ---- write C: acc VGPR v -> row (v + 8*laneHi), lane&15 -> col ----
    float* Cb = C + (size_t)batch * Nrows * Mcols;
    const int rowBase = nBase + waveM * 64 + (laneHi << 3);
    const int colBase = mBase + waveN * 32 + r16;
#pragma unroll
    for (int tm = 0; tm < 4; ++tm)
#pragma unroll
        for (int tn = 0; tn < 2; ++tn)
#pragma unroll
            for (int v = 0; v < 8; ++v) {
                int row = rowBase + tm * 16 + v;
                int col = colBase + tn * 16;
                Cb[(size_t)row * Mcols + col] = acc[tm][tn][v];
            }
}

extern "C" void kernel_launch(void* const* d_in, const int* in_sizes, int n_in,
                              void* d_out, int out_size, void* d_ws, size_t ws_size,
                              hipStream_t stream) {
    (void)in_sizes; (void)n_in; (void)d_ws; (void)ws_size; (void)out_size;
    const int b = 8, N = 2048, M = 2048, E = 1024;
    const float* mat0 = (const float*)d_in[0];
    const float* mat1 = (const float*)d_in[1];
    float* out = (float*)d_out;

    dim3 grid(M / BN, N / BM, b);   // (16, 16, 8)
    dim3 block(NT);                 // 8 wave32
    attn_qkt_bf16x3_kernel<<<grid, block, 0, stream>>>(mat0, mat1, out, N, M, E);
}
